// LocalEnergy_2628519985582
// MI455X (gfx1250) — matrix operand order repaired and checked by reading, not compiled, not run on hardware
//
#include <hip/hip_runtime.h>

typedef __attribute__((ext_vector_type(16))) _Float16 v16h;
typedef __attribute__((ext_vector_type(8)))  float    v8f;

#define BB 128
#define LL 2048
#define HH 128
#define BUFC 136            // halves per row in per-wave LDS buffer (128 + pad)
#define NTILES 16           // 16 tiles of 128 rows cover up to 2048 rows
#define TPW 2               // tiles per workgroup (grid.x = NTILES/TPW)

__global__ void zero_out_kernel(float* out) {
    out[threadIdx.x] = 0.0f;
}

// MODE 0: bond-length MLP  d=33  (1 scalar + 2 embeddings), K padded to 64
// MODE 1: angle MLP        d=49  (1 scalar + 3 embeddings), K padded to 64
// MODE 2: dihedral MLP     d=66  (2 scalars + 4 embeddings), K padded to 96
template<int MODE>
__global__ __launch_bounds__(256)
void mlp_energy_kernel(const float* __restrict__ R, const int* __restrict__ seq,
                       const float* __restrict__ emb,
                       const float* __restrict__ W1, const float* __restrict__ B1v,
                       const float* __restrict__ W2, const float* __restrict__ B2v,
                       const float* __restrict__ W3, const float* __restrict__ B3v,
                       float* __restrict__ out)
{
    constexpr int SC   = (MODE == 2) ? 2 : 1;     // scalar feature count
    constexpr int NE   = MODE + 2;                // embedding vectors
    constexpr int D    = SC + NE * 16;            // true input dim
    constexpr int KPAD = ((D + 31) / 32) * 32;    // padded K for 16x16x32 WMMA
    constexpr int NK1  = KPAD / 32;               // K-steps in layer 1
    constexpr int ROWS = LL - 1 - MODE;           // valid rows per batch

    extern __shared__ __attribute__((aligned(32))) char smem[];
    _Float16* sW1f = (_Float16*)smem;             // KPAD*128 halves, frag-packed
    _Float16* sW2f = sW1f + KPAD * HH;            // 128*128 halves, frag-packed
    float*    sB1  = (float*)(sW2f + HH * HH);    // 128
    float*    sB2  = sB1 + HH;                    // 128
    float*    sW3  = sB2 + HH;                    // 128
    _Float16* sBuf = (_Float16*)(sW3 + HH);       // 8 waves * 16 rows * BUFC

    const int tid  = threadIdx.x;
    const int wave = tid >> 5;
    const int lane = tid & 31;
    const int m16  = lane & 15;
    const int g    = lane >> 4;
    const int b    = blockIdx.y;

    // ---- stage weights into fragment-packed f16 LDS layout ----
    // B-fragment packing used here (contiguous k, same order as A): lane = (N%16) | ((K%32)/16)<<4 ; j = K%16
    for (int idx = tid; idx < KPAD * HH; idx += 256) {
        int K = idx >> 7, N = idx & (HH - 1);
        float v = (K < D) ? W1[K * HH + N] : 0.0f;
        int k = K >> 5, n = N >> 4;
        int dl = (N & 15) | (((K >> 4) & 1) << 4);
        sW1f[((n * NK1 + k) * 32 + dl) * 16 + (K & 15)] = (_Float16)v;
    }
    for (int idx = tid; idx < HH * HH; idx += 256) {
        int K = idx >> 7, N = idx & (HH - 1);
        float v = W2[K * HH + N];
        int k = K >> 5, n = N >> 4;
        int dl = (N & 15) | (((K >> 4) & 1) << 4);
        sW2f[((n * 4 + k) * 32 + dl) * 16 + (K & 15)] = (_Float16)v;
    }
    if (tid < HH) { sB1[tid] = B1v[tid]; sB2[tid] = B2v[tid]; sW3[tid] = W3[tid]; }
    __syncthreads();

    const float b3f = B3v[0];
    _Float16* myBuf = sBuf + wave * 16 * BUFC;

    for (int t = 0; t < TPW; ++t) {
        const int tile = blockIdx.x + t * (NTILES / TPW);
        const int rowBase = tile * 128 + wave * 16;

        // ---- per-row features (lanes 0..15, one row each) ----
        if (lane < 16) {
            int l  = rowBase + lane;
            int lc = (l < ROWS) ? l : (ROWS - 1);   // clamp: masked later
            const float* rp = R + ((size_t)b * LL + lc) * 3;
            _Float16* xr = myBuf + lane * BUFC;
            if (MODE == 0) {
                float dx = rp[3] - rp[0], dy = rp[4] - rp[1], dz = rp[5] - rp[2];
                xr[0] = (_Float16)sqrtf(dx * dx + dy * dy + dz * dz);
            } else if (MODE == 1) {
                float ux = rp[0] - rp[3], uy = rp[1] - rp[4], uz = rp[2] - rp[5];
                float vx = rp[6] - rp[3], vy = rp[7] - rp[4], vz = rp[8] - rp[5];
                float dot = ux * vx + uy * vy + uz * vz;
                float nu = sqrtf(ux * ux + uy * uy + uz * uz);
                float nv = sqrtf(vx * vx + vy * vy + vz * vz);
                float c = dot / (nu * nv);
                c = fminf(1.0f, fmaxf(-1.0f, c));
                xr[0] = (_Float16)c;
            } else {
                float b1x = rp[3] - rp[0], b1y = rp[4] - rp[1], b1z = rp[5] - rp[2];
                float b2x = rp[6] - rp[3], b2y = rp[7] - rp[4], b2z = rp[8] - rp[5];
                float b3x = rp[9] - rp[6], b3y = rp[10] - rp[7], b3z = rp[11] - rp[8];
                float n1x = b1y * b2z - b1z * b2y;
                float n1y = b1z * b2x - b1x * b2z;
                float n1z = b1x * b2y - b1y * b2x;
                float n2x = b2y * b3z - b2z * b3y;
                float n2y = b2z * b3x - b2x * b3z;
                float n2z = b2x * b3y - b2y * b3x;
                float ib2 = 1.0f / sqrtf(b2x * b2x + b2y * b2y + b2z * b2z);
                float ux = b2x * ib2, uy = b2y * ib2, uz = b2z * ib2;
                float m1x = n1y * uz - n1z * uy;
                float m1y = n1z * ux - n1x * uz;
                float m1z = n1x * uy - n1y * ux;
                float yv = m1x * n2x + m1y * n2y + m1z * n2z;  // -|b2|*dot(b1,b2xb3) scale-free after norm
                float xv = n1x * n2x + n1y * n2y + n1z * n2z;
                float r = fmaxf(sqrtf(xv * xv + yv * yv), 1e-20f);
                xr[0] = (_Float16)(yv / r);   // sin(phi)
                xr[1] = (_Float16)(xv / r);   // cos(phi)
            }
            const int* sp = seq + (size_t)b * LL + lc;
            #pragma unroll
            for (int tt = 0; tt < NE; ++tt) {
                const float* ep = emb + sp[tt] * 16;
                #pragma unroll
                for (int c = 0; c < 16; ++c)
                    xr[SC + tt * 16 + c] = (_Float16)ep[c];
            }
            #pragma unroll
            for (int K = D; K < KPAD; ++K) xr[K] = (_Float16)0.0f;
        }

        // ---- layer 1: x[16xKPAD] @ W1[KPADx128] ----
        // A fragment packing used here (contiguous k, same order as B): lane m=L%16; half j: K = (L/16)*16 + j
        v16h a1[NK1];
        const _Float16* arp = myBuf + m16 * BUFC;
        #pragma unroll
        for (int k = 0; k < NK1; ++k)
            #pragma unroll
            for (int jj = 0; jj < 2; ++jj)
                #pragma unroll
                for (int q = 0; q < 8; ++q)
                    a1[k][jj * 8 + q] = arp[k * 32 + g * 16 + jj * 8 + q];

        #pragma unroll
        for (int n = 0; n < 8; ++n) {
            float bias1 = sB1[n * 16 + m16];
            v8f acc;
            #pragma unroll
            for (int v = 0; v < 8; ++v) acc[v] = bias1;
            #pragma unroll
            for (int k = 0; k < NK1; ++k) {
                v16h bf = *(const v16h*)(sW1f + ((n * NK1 + k) * 32 + lane) * 16);
                acc = __builtin_amdgcn_wmma_f32_16x16x32_f16(
                        false, a1[k], false, bf, (short)0, acc, false, false);
            }
            // ReLU + write h back: C element (m = v + 8g, col = n*16 + m16)
            int col = n * 16 + m16;
            #pragma unroll
            for (int v = 0; v < 8; ++v)
                myBuf[(v + 8 * g) * BUFC + col] = (_Float16)fmaxf(acc[v], 0.0f);
        }

        // ---- layer 2: h[16x128] @ W2[128x128], fused with layer 3 dot(W3) ----
        v16h a2[4];
        #pragma unroll
        for (int k = 0; k < 4; ++k)
            #pragma unroll
            for (int jj = 0; jj < 2; ++jj)
                #pragma unroll
                for (int q = 0; q < 8; ++q)
                    a2[k][jj * 8 + q] = arp[k * 32 + g * 16 + jj * 8 + q];

        float sv[8];
        #pragma unroll
        for (int v = 0; v < 8; ++v) sv[v] = 0.0f;

        #pragma unroll
        for (int n = 0; n < 8; ++n) {
            float bias2 = sB2[n * 16 + m16];
            v8f acc;
            #pragma unroll
            for (int v = 0; v < 8; ++v) acc[v] = bias2;
            #pragma unroll
            for (int k = 0; k < 4; ++k) {
                v16h bf = *(const v16h*)(sW2f + ((n * 4 + k) * 32 + lane) * 16);
                acc = __builtin_amdgcn_wmma_f32_16x16x32_f16(
                        false, a2[k], false, bf, (short)0, acc, false, false);
            }
            float w3v = sW3[n * 16 + m16];
            #pragma unroll
            for (int v = 0; v < 8; ++v)
                sv[v] += fmaxf(acc[v], 0.0f) * w3v;
        }

        // ---- mask invalid rows, add per-row b3, reduce wave, accumulate ----
        float s = 0.0f;
        #pragma unroll
        for (int v = 0; v < 8; ++v) {
            int m = v + 8 * g;
            if (rowBase + m < ROWS) {
                s += sv[v];
                if (m16 == 0) s += b3f;   // b3 once per valid row
            }
        }
        #pragma unroll
        for (int off = 16; off > 0; off >>= 1)
            s += __shfl_xor(s, off, 32);
        if (lane == 0) atomicAdd(out + b, s);
    }
}

extern "C" void kernel_launch(void* const* d_in, const int* in_sizes, int n_in,
                              void* d_out, int out_size, void* d_ws, size_t ws_size,
                              hipStream_t stream) {
    const float* R    = (const float*)d_in[0];
    const int*   seq  = (const int*)  d_in[1];
    const float* emb  = (const float*)d_in[2];
    const float* flW1 = (const float*)d_in[3];
    const float* flb1 = (const float*)d_in[4];
    const float* flW2 = (const float*)d_in[5];
    const float* flb2 = (const float*)d_in[6];
    const float* flW3 = (const float*)d_in[7];
    const float* flb3 = (const float*)d_in[8];
    const float* ftW1 = (const float*)d_in[9];
    const float* ftb1 = (const float*)d_in[10];
    const float* ftW2 = (const float*)d_in[11];
    const float* ftb2 = (const float*)d_in[12];
    const float* ftW3 = (const float*)d_in[13];
    const float* ftb3 = (const float*)d_in[14];
    const float* fpW1 = (const float*)d_in[15];
    const float* fpb1 = (const float*)d_in[16];
    const float* fpW2 = (const float*)d_in[17];
    const float* fpb2 = (const float*)d_in[18];
    const float* fpW3 = (const float*)d_in[19];
    const float* fpb3 = (const float*)d_in[20];
    float* out = (float*)d_out;

    // dynamic LDS: W1f + W2f + (b1,b2,w3) + 8-wave row buffer
    const int smem01 = 64 * 128 * 2 + 128 * 128 * 2 + 3 * 128 * 4 + 8 * 16 * BUFC * 2; // 85504
    const int smem2  = 96 * 128 * 2 + 128 * 128 * 2 + 3 * 128 * 4 + 8 * 16 * BUFC * 2; // 93696
    (void)hipFuncSetAttribute((const void*)mlp_energy_kernel<0>,
                              hipFuncAttributeMaxDynamicSharedMemorySize, smem01);
    (void)hipFuncSetAttribute((const void*)mlp_energy_kernel<1>,
                              hipFuncAttributeMaxDynamicSharedMemorySize, smem01);
    (void)hipFuncSetAttribute((const void*)mlp_energy_kernel<2>,
                              hipFuncAttributeMaxDynamicSharedMemorySize, smem2);

    zero_out_kernel<<<1, 128, 0, stream>>>(out);

    dim3 grid(NTILES / TPW, BB);
    mlp_energy_kernel<0><<<grid, 256, smem01, stream>>>(
        R, seq, emb, flW1, flb1, flW2, flb2, flW3, flb3, out);
    mlp_energy_kernel<1><<<grid, 256, smem01, stream>>>(
        R, seq, emb, ftW1, ftb1, ftW2, ftb2, ftW3, ftb3, out);
    mlp_energy_kernel<2><<<grid, 256, smem2, stream>>>(
        R, seq, emb, fpW1, fpb1, fpW2, fpb2, fpW3, fpb3, out);
}
